// STGCNBlock_90520730730508
// MI455X (gfx1250) — matrix operand order repaired and checked by reading, not compiled, not run on hardware
//
#include <hip/hip_runtime.h>

typedef __attribute__((ext_vector_type(16))) _Float16 v16h;
typedef __attribute__((ext_vector_type(8)))  float    v8f;

#define CIN  32
#define COUT 64
#define S_T  12

// Load 8 consecutive f32 (16B aligned) and convert into halves [base, base+8) of a v16h.
__device__ __forceinline__ void load8_h(const float* __restrict__ p, v16h& d, const int base) {
  float4 f0 = ((const float4*)p)[0];
  float4 f1 = ((const float4*)p)[1];
  d[base + 0] = (_Float16)f0.x; d[base + 1] = (_Float16)f0.y;
  d[base + 2] = (_Float16)f0.z; d[base + 3] = (_Float16)f0.w;
  d[base + 4] = (_Float16)f1.x; d[base + 5] = (_Float16)f1.y;
  d[base + 6] = (_Float16)f1.z; d[base + 7] = (_Float16)f1.w;
}

// ---------------- degree / dinv ----------------
__global__ void deg_init_kernel(float* __restrict__ deg, int N) {
  int i = blockIdx.x * blockDim.x + threadIdx.x;
  if (i < N) deg[i] = 1.0f;  // self-loop
}

__global__ void deg_scatter_kernel(const int* __restrict__ ei, float* __restrict__ deg, int E) {
  int e = blockIdx.x * blockDim.x + threadIdx.x;
  if (e < E) atomicAdd(&deg[ei[E + e]], 1.0f);  // dst row of edge_index
}

__global__ void dinv_kernel(const float* __restrict__ deg, float* __restrict__ dinv, int N) {
  int i = blockIdx.x * blockDim.x + threadIdx.x;
  if (i < N) {
    float d = deg[i];
    dinv[i] = d > 0.0f ? rsqrtf(d) : 0.0f;
  }
}

// ---------------- fused tconv + residual (WMMA) ----------------
// One wave per 16-row tile of the [S*N, CIN] input. Shared A fragment feeds two
// GEMMs (tconv weights -> h with BN+ReLU, residual weights -> out).
__global__ void tconv_res_kernel(const float* __restrict__ x,
                                 const float* __restrict__ tw, const float* __restrict__ tb,
                                 const float* __restrict__ tg, const float* __restrict__ tbe,
                                 const float* __restrict__ rw, const float* __restrict__ rb,
                                 float* __restrict__ h, float* __restrict__ out, int SN) {
  int wave = (blockIdx.x * blockDim.x + threadIdx.x) >> 5;
  int lane = threadIdx.x & 31;
  int numTiles = SN >> 4;
  if (wave >= numTiles) return;  // whole-wave uniform: EXEC stays all-1s for WMMA

  const float inv = rsqrtf(1.0f + 1e-5f);
  int rowBase = wave << 4;
  int m       = lane & 15;
  bool hi     = lane >= 16;
  int kA      = hi ? 8 : 0;    // unused; A is loaded with kB (k = 16h + s)
  int kB      = hi ? 16 : 0;   // contiguous K order k = 16h + s, used for both A and B
  int mrow    = hi ? 8 : 0;    // C/D fragment row base

  const float* xrow = x + (size_t)(rowBase + m) * CIN;
  v16h a;
  load8_h(xrow + kB,      a, 0);
  load8_h(xrow + kB + 8,  a, 8);

#pragma unroll
  for (int ct = 0; ct < 4; ++ct) {
    int oc = ct * 16 + (lane & 15);
    v16h bt, br;
    load8_h(tw + (size_t)oc * CIN + kB,     bt, 0);
    load8_h(tw + (size_t)oc * CIN + kB + 8, bt, 8);
    load8_h(rw + (size_t)oc * CIN + kB,     br, 0);
    load8_h(rw + (size_t)oc * CIN + kB + 8, br, 8);

    v8f c0 = {}; v8f c1 = {};
    c0 = __builtin_amdgcn_wmma_f32_16x16x32_f16(false, a, false, bt, (short)0, c0, false, false);
    c1 = __builtin_amdgcn_wmma_f32_16x16x32_f16(false, a, false, br, (short)0, c1, false, false);

    float tbv = tb[oc];
    float tgv = tg[oc] * inv;
    float tbb = tbe[oc];
    float rbv = rb[oc];
#pragma unroll
    for (int v = 0; v < 8; ++v) {
      size_t rr = (size_t)(rowBase + mrow + v);
      float hv = (c0[v] + tbv) * tgv + tbb;
      h[rr * COUT + oc]   = hv > 0.0f ? hv : 0.0f;
      out[rr * COUT + oc] = c1[v] + rbv;  // residual; GCN output accumulated later
    }
  }
}

// ---------------- GCN linear per timestep (WMMA, K=64 in two steps) ----------------
__global__ void gcn_lin_kernel(const float* __restrict__ h_t, const float* __restrict__ gw,
                               float* __restrict__ ht, int N) {
  int wave = (blockIdx.x * blockDim.x + threadIdx.x) >> 5;
  int lane = threadIdx.x & 31;
  int numTiles = N >> 4;
  if (wave >= numTiles) return;

  int rowBase = wave << 4;
  int m       = lane & 15;
  bool hi     = lane >= 16;
  int kA      = hi ? 8 : 0;
  int kB      = hi ? 16 : 0;
  int mrow    = hi ? 8 : 0;

  const float* hrow = h_t + (size_t)(rowBase + m) * COUT;
  v16h a0, a1;
  load8_h(hrow + kB,          a0, 0);
  load8_h(hrow + kB + 8,      a0, 8);
  load8_h(hrow + 32 + kB,     a1, 0);
  load8_h(hrow + 32 + kB + 8, a1, 8);

#pragma unroll
  for (int ct = 0; ct < 4; ++ct) {
    int oc = ct * 16 + (lane & 15);
    const float* wrow = gw + (size_t)oc * COUT;
    v16h b0, b1;
    load8_h(wrow + kB,          b0, 0);
    load8_h(wrow + kB + 8,      b0, 8);
    load8_h(wrow + 32 + kB,     b1, 0);
    load8_h(wrow + 32 + kB + 8, b1, 8);

    v8f c = {};
    c = __builtin_amdgcn_wmma_f32_16x16x32_f16(false, a0, false, b0, (short)0, c, false, false);
    c = __builtin_amdgcn_wmma_f32_16x16x32_f16(false, a1, false, b1, (short)0, c, false, false);
#pragma unroll
    for (int v = 0; v < 8; ++v)
      ht[(size_t)(rowBase + mrow + v) * COUT + oc] = c[v];
  }
}

// ---------------- message passing ----------------
__global__ void agg_init_kernel(const float* __restrict__ ht, const float* __restrict__ dinv,
                                float* __restrict__ agg, int N) {
  int i = blockIdx.x * blockDim.x + threadIdx.x;
  if (i < N * COUT) {
    int n = i >> 6;
    float dv = dinv[n];
    agg[i] = ht[i] * dv * dv;  // self-loop contribution
  }
}

__global__ void edge_scatter_kernel(const int* __restrict__ ei, const float* __restrict__ ht,
                                    const float* __restrict__ dinv, float* __restrict__ agg, int E) {
  long long i = (long long)blockIdx.x * blockDim.x + threadIdx.x;
  if (i >= (long long)E * COUT) return;
  int c = (int)(i & 63);
  int e = (int)(i >> 6);
  int s = ei[e];
  int d = ei[E + e];
  float nrm = dinv[s] * dinv[d];
  atomicAdd(&agg[(size_t)d * COUT + c], ht[(size_t)s * COUT + c] * nrm);
}

__global__ void epilogue_kernel(const float* __restrict__ agg, const float* __restrict__ gb,
                                const float* __restrict__ sg, const float* __restrict__ sb,
                                float* __restrict__ out_t, int N) {
  int i = blockIdx.x * blockDim.x + threadIdx.x;
  if (i < N * COUT) {
    const float inv = rsqrtf(1.0f + 1e-5f);
    int c = i & 63;
    float v = (agg[i] + gb[c]) * (sg[c] * inv) + sb[c];
    out_t[i] += v > 0.0f ? v : 0.0f;
  }
}

extern "C" void kernel_launch(void* const* d_in, const int* in_sizes, int n_in,
                              void* d_out, int out_size, void* d_ws, size_t ws_size,
                              hipStream_t stream) {
  const float* x    = (const float*)d_in[0];
  const float* tw   = (const float*)d_in[1];
  const float* tb   = (const float*)d_in[2];
  const float* tg   = (const float*)d_in[3];
  const float* tbe  = (const float*)d_in[4];
  const float* gw   = (const float*)d_in[5];
  const float* gb   = (const float*)d_in[6];
  const float* sg   = (const float*)d_in[7];
  const float* sb   = (const float*)d_in[8];
  const float* rw   = (const float*)d_in[9];
  const float* rb   = (const float*)d_in[10];
  const int*   ei   = (const int*)d_in[11];
  float*       out  = (float*)d_out;

  const int SN = in_sizes[0] / CIN;   // 600000
  const int N  = SN / S_T;            // 50000
  const int E  = in_sizes[11] / 2;    // 800000

  // workspace layout (floats)
  float* ws   = (float*)d_ws;
  float* deg  = ws;                          // N
  float* dinv = deg + N;                     // N
  float* h    = dinv + N;                    // SN*COUT
  float* ht   = h + (size_t)SN * COUT;       // N*COUT
  float* agg  = ht + (size_t)N * COUT;       // N*COUT

  const int TB = 256;

  // degrees -> dinv
  deg_init_kernel<<<(N + TB - 1) / TB, TB, 0, stream>>>(deg, N);
  deg_scatter_kernel<<<(E + TB - 1) / TB, TB, 0, stream>>>(ei, deg, E);
  dinv_kernel<<<(N + TB - 1) / TB, TB, 0, stream>>>(deg, dinv, N);

  // fused temporal conv + residual (WMMA): one wave per 16-row tile
  {
    int tiles  = SN >> 4;                       // 37500
    int waves  = TB / 32;                       // 8 waves / block
    int blocks = (tiles + waves - 1) / waves;
    tconv_res_kernel<<<blocks, TB, 0, stream>>>(x, tw, tb, tg, tbe, rw, rb, h, out, SN);
  }

  // per-timestep GCN
  for (int t = 0; t < S_T; ++t) {
    const float* h_t  = h + (size_t)t * N * COUT;
    float*       outt = out + (size_t)t * N * COUT;

    int tiles  = N >> 4;                        // 3125
    int waves  = TB / 32;
    int blocks = (tiles + waves - 1) / waves;
    gcn_lin_kernel<<<blocks, TB, 0, stream>>>(h_t, gw, ht, N);

    int nc = N * COUT;
    agg_init_kernel<<<(nc + TB - 1) / TB, TB, 0, stream>>>(ht, dinv, agg, N);

    long long ec = (long long)E * COUT;
    int eblocks  = (int)((ec + TB - 1) / TB);
    edge_scatter_kernel<<<eblocks, TB, 0, stream>>>(ei, ht, dinv, agg, E);

    epilogue_kernel<<<(nc + TB - 1) / TB, TB, 0, stream>>>(agg, gb, sg, sb, outt, N);
  }
}
